// Discriminator_28501402976265
// MI455X (gfx1250) — matrix operand order repaired and checked by reading, not compiled, not run on hardware
//
#include <hip/hip_runtime.h>
#include <hip/hip_bf16.h>

typedef __attribute__((ext_vector_type(16))) __bf16 v16bf;
typedef __attribute__((ext_vector_type(8)))  float  v8f;

#define N_NODES 50000
#define IN_DIM  512
#define HID     256
#define E_EDGES 800000
#define TEMP_   1.0f
#define BIAS_   0.0001f

#define ROW_TILES (N_NODES / 16)   // 3125
#define COL_TILES (HID / 16)       // 16
#define K_STEPS   (IN_DIM / 32)    // 16
#define WFRAG_ELEMS (COL_TILES * K_STEPS * 32 * 16)  // 131072 bf16 = 256KB
#define AGROUPS   (K_STEPS * 2)    // 32 groups of 32 lane-slots (16B each)
#define AGPITCH   33               // pad 1 uint4 per group: conflict-free stores

__device__ __forceinline__ unsigned int pack_bf16(float a, float b) {
    union { __bf16 h[2]; unsigned int u; } r;
    r.h[0] = (__bf16)a; r.h[1] = (__bf16)b;
    return r.u;   // lowers to v_cvt_pk_bf16_f32
}

// ---------------------------------------------------------------------------
// Kernel 0: repack W_emb (f32 [512][256] row-major) into bf16 B-fragments in
// wave32 WMMA layout: lanes 0-15 hold column N=lane%16, K 0..7,16..23 of the k-step;
// lanes 16-31 hold K 8..15,24..31.  Linear layout: [colTile][kstep][lane][16].
// ---------------------------------------------------------------------------
__global__ void prep_wfrag(const float* __restrict__ W, __bf16* __restrict__ wfrag) {
    int tid = blockIdx.x * blockDim.x + threadIdx.x;
    if (tid >= WFRAG_ELEMS) return;
    int j       = tid & 15;
    int lane    = (tid >> 4) & 31;
    int kstep   = (tid >> 9) & 15;
    int colTile = tid >> 13;
    int col = colTile * 16 + (lane & 15);
    int k   = kstep * 32 + (lane >> 4) * 8 + (j & 7) + (j >> 3) * 16;
    wfrag[tid] = (__bf16)W[k * HID + col];
}

// ---------------------------------------------------------------------------
// Kernel 1: h = relu(emb @ W + b).  One block = one 16-row tile; 16 waves
// cover the 16 column tiles (HID=256).  A tile is staged into LDS *already
// converted to bf16 and already in per-lane WMMA fragment layout*, so the
// K-loop is just 2 ds_load_b128 + 2 global_load_b128 + 1 v_wmma per step.
//
// A-frag layout (ISA 16-bit A, 16x32): lane<16 holds row M=lane,
// elems 0..7 = K kc+0..7, elems 8..15 = K kc+16..23; lane>=16 holds row
// M=lane-16, elems 0..7 = K kc+8..15, elems 8..15 = K kc+24..31.
// LDS: group g = kstep*2 + half (half 0: elems 0..7, half 1: elems 8..15),
// slot = lane; 16B per slot, groups padded to 33 uint4.
// ---------------------------------------------------------------------------
__global__ void __launch_bounds__(512)
gemm_relu(const float* __restrict__ emb, const __bf16* __restrict__ wfrag,
          const float* __restrict__ b_emb, float* __restrict__ h) {
    __shared__ uint4 asmem[AGROUPS * AGPITCH];   // 16.9 KB
    const int t    = threadIdx.x;
    const int wave = t >> 5;          // column tile 0..15
    const int lane = t & 31;
    const int rowBase = blockIdx.x * 16;

    // Stage + convert: 1024 chunks of 8 consecutive K per row; 2 per thread.
    // Global reads: 32B contiguous per thread, coalesced across the wave.
#pragma unroll
    for (int i = 0; i < 2; ++i) {
        const int c     = i * 512 + t;   // 0..1023
        const int m     = c >> 6;        // row in tile
        const int ch    = c & 63;        // 8-float chunk within row
        const int k0    = ch << 3;
        const int kstep = k0 >> 5;
        const int kin0  = k0 & 31;       // 0, 8, 16, 24
        const int khalf = (kin0 >> 3) & 1;
        const int half  = kin0 >> 4;
        const int slot  = m + (khalf << 4);
        const float4* src = (const float4*)(emb + (size_t)(rowBase + m) * IN_DIM + k0);
        float4 f0 = src[0], f1 = src[1];
        uint4 q;
        q.x = pack_bf16(f0.x, f0.y);
        q.y = pack_bf16(f0.z, f0.w);
        q.z = pack_bf16(f1.x, f1.y);
        q.w = pack_bf16(f1.z, f1.w);
        asmem[(kstep * 2 + half) * AGPITCH + slot] = q;
    }
    __syncthreads();

    v8f c = {};
    const v16bf* wf = (const v16bf*)wfrag + (size_t)wave * K_STEPS * 32;

#pragma unroll
    for (int ks = 0; ks < K_STEPS; ++ks) {
        union { uint4 q[2]; v16bf v; } av;
        av.q[0] = asmem[(ks * 2 + 0) * AGPITCH + lane];
        av.q[1] = asmem[(ks * 2 + 1) * AGPITCH + lane];
        v16bf b = wf[ks * 32 + lane];
        c = __builtin_amdgcn_wmma_f32_16x16x32_bf16(
                /*neg_a=*/false, av.v, /*neg_b=*/false, b,
                /*c_mod=*/(short)0, c, /*reuse_a=*/false, /*reuse_b=*/false);
    }

    // C layout: VGPR r, lane l -> M = r + (l/16)*8, N = l%16. Fuse bias + ReLU.
    const int khalfc = lane >> 4;
    const int col    = wave * 16 + (lane & 15);
    const float bv   = b_emb[col];
#pragma unroll
    for (int r = 0; r < 8; ++r) {
        const int m = r + khalfc * 8;
        float v = c[r] + bv;
        h[(size_t)(rowBase + m) * HID + col] = v > 0.0f ? v : 0.0f;
    }
}

// ---------------------------------------------------------------------------
// Kernel 2: per-edge symmetric dot + gumbel-sigmoid gate.  One wave32 per
// edge: lane covers 8 of the 256 channels (2x float4), shfl-xor reduction.
// h (51.2MB) is L2-resident (192MB L2), so gathers run at L2 speed.
// ---------------------------------------------------------------------------
__global__ void __launch_bounds__(256)
edge_gate(const float* __restrict__ h, const long long* __restrict__ edges,
          const float* __restrict__ u, const float* __restrict__ W_edge,
          const float* __restrict__ b_edge, float* __restrict__ out) {
    const int wave = threadIdx.x >> 5;
    const int lane = threadIdx.x & 31;
    const long long e  = (long long)blockIdx.x * 8 + wave;
    const long long i0 = edges[e];
    const long long i1 = edges[(long long)E_EDGES + e];

    const float4* h0 = (const float4*)(h + (size_t)i0 * HID) + lane * 2;
    const float4* h1 = (const float4*)(h + (size_t)i1 * HID) + lane * 2;
    const float4* we = (const float4*)W_edge + lane * 2;   // we[q+64] = second half

    float s = 0.0f;
#pragma unroll
    for (int q = 0; q < 2; ++q) {
        float4 a = h0[q], b = h1[q];
        float4 wA = we[q], wB = we[q + 64];
        s += (a.x + b.x) * 0.5f * (wA.x + wB.x);
        s += (a.y + b.y) * 0.5f * (wA.y + wB.y);
        s += (a.z + b.z) * 0.5f * (wA.z + wB.z);
        s += (a.w + b.w) * 0.5f * (wA.w + wB.w);
    }
#pragma unroll
    for (int off = 16; off > 0; off >>= 1)
        s += __shfl_xor(s, off, 32);

    if (lane == 0) {
        const float raw  = s + b_edge[0];
        const float uu   = u[e];
        const float eps  = (1.0f - BIAS_) - (1.0f - 2.0f * BIAS_) * uu;
        const float gate = (logf(eps) - log1pf(-eps) + raw) / TEMP_;
        out[e] = 1.0f / (1.0f + expf(-gate));
    }
}

// ---------------------------------------------------------------------------
extern "C" void kernel_launch(void* const* d_in, const int* in_sizes, int n_in,
                              void* d_out, int out_size, void* d_ws, size_t ws_size,
                              hipStream_t stream) {
    const float*     emb    = (const float*)d_in[0];
    const long long* edges  = (const long long*)d_in[1];
    const float*     u      = (const float*)d_in[2];
    const float*     W_emb  = (const float*)d_in[3];
    const float*     b_emb  = (const float*)d_in[4];
    const float*     W_edge = (const float*)d_in[5];
    const float*     b_edge = (const float*)d_in[6];
    float*           out    = (float*)d_out;

    __bf16* wfrag = (__bf16*)d_ws;                                       // 256 KB
    float*  h     = (float*)((char*)d_ws + WFRAG_ELEMS * sizeof(__bf16)); // 51.2 MB

    prep_wfrag<<<WFRAG_ELEMS / 256, 256, 0, stream>>>(W_emb, wfrag);
    gemm_relu<<<ROW_TILES, 512, 0, stream>>>(emb, wfrag, b_emb, h);
    edge_gate<<<E_EDGES / 8, 256, 0, stream>>>(h, edges, u, W_edge, b_edge, out);
}
